// GraphTransformerLayer_11768210391529
// MI455X (gfx1250) — matrix operand order repaired and checked by reading, not compiled, not run on hardware
//
#include <hip/hip_runtime.h>
#include <hip/hip_bf16.h>

#define NN  2048
#define DD  128
#define QKD 64
#define HH  8
#define EE  8

typedef __attribute__((ext_vector_type(16))) __bf16 v16bf;
typedef __attribute__((ext_vector_type(2)))  __bf16 v2bf;
typedef __attribute__((ext_vector_type(8)))  float  v8f;
typedef __attribute__((ext_vector_type(4)))  float  v4f;

union BFV {
    v16bf v;
    unsigned short u16[16];
    unsigned u32[8];
    uint4 q[2];
};

union BF2U { v2bf v; unsigned u; };

// Pack two floats into packed bf16 (RNE) — hw converter.
__device__ __forceinline__ unsigned f2bf2(float lo, float hi) {
#if __has_builtin(__builtin_amdgcn_cvt_pk_bf16_f32)
    BF2U b; b.v = __builtin_amdgcn_cvt_pk_bf16_f32(lo, hi);
    return b.u;
#else
    unsigned r;
    asm("v_cvt_pk_bf16_f32 %0, %1, %2" : "=v"(r) : "v"(lo), "v"(hi));
    return r;
#endif
}

__device__ __forceinline__ unsigned short f2bf(float f) {
    return (unsigned short)(f2bf2(f, f) & 0xFFFFu);
}

__device__ __forceinline__ v8f wmma_bf16(const BFV& a, const BFV& b, v8f c) {
    return __builtin_amdgcn_wmma_f32_16x16x32_bf16(false, a.v, false, b.v,
                                                   (short)0, c, false, false);
}

__device__ __forceinline__ void pack8(BFV& d, int w, float4 a, float4 b) {
    d.u32[w + 0] = f2bf2(a.x, a.y);
    d.u32[w + 1] = f2bf2(a.z, a.w);
    d.u32[w + 2] = f2bf2(b.x, b.y);
    d.u32[w + 3] = f2bf2(b.z, b.w);
}

__device__ __forceinline__ void atomicMaxF(float* addr, float val) {
    unsigned int* p = (unsigned int*)addr;
    unsigned int cur = *p;
    while (__uint_as_float(cur) < val) {
        unsigned int prev = atomicCAS(p, cur, __float_as_uint(val));
        if (prev == cur) break;
        cur = prev;
    }
}

__device__ __forceinline__ void atomicMinF(float* addr, float val) {
    unsigned int* p = (unsigned int*)addr;
    unsigned int cur = *p;
    while (__uint_as_float(cur) > val) {
        unsigned int prev = atomicCAS(p, cur, __float_as_uint(val));
        if (prev == cur) break;
        cur = prev;
    }
}

// Asynchronously stage one 16x16 (edges 8-vec + adj) tile into LDS.
// 3 async ops per wave, tracked on ASYNCcnt (no VGPR round-trip).
__device__ __forceinline__ void stage_tile_async(const float* __restrict__ eg,
                                                 const float* __restrict__ ag,
                                                 float* seb, float* adjb, int t) {
    unsigned lds_e = (unsigned)(size_t)(seb + t * 8);
    asm volatile("global_load_async_to_lds_b128 %0, %1, off"
                 :: "v"(lds_e), "v"(eg) : "memory");
    asm volatile("global_load_async_to_lds_b128 %0, %1, off offset:16"
                 :: "v"(lds_e), "v"(eg) : "memory");
    unsigned lds_a = (unsigned)(size_t)(adjb + t);
    asm volatile("global_load_async_to_lds_b32 %0, %1, off"
                 :: "v"(lds_a), "v"(ag) : "memory");
}

// ---------------------------------------------------------------------------
// Pass 0: zero / init accumulators in workspace (ws is poisoned, not re-zeroed)
// ---------------------------------------------------------------------------
__global__ void __launch_bounds__(256) k_init(float* __restrict__ hacc,
                                              float* __restrict__ rowsum,
                                              float* __restrict__ rowmax,
                                              float* __restrict__ gmin) {
    int i = blockIdx.x * blockDim.x + threadIdx.x;
    if (i < HH * NN * QKD) hacc[i] = 0.f;
    if (i < HH * NN) { rowsum[i] = 0.f; rowmax[i] = -3.0e38f; }
    if (i == 0) gmin[0] = 3.0e38f;
}

// ---------------------------------------------------------------------------
// Pass 1: QKV projection via WMMA bf16.  q,k stored [h][n][d] bf16; v stored
// transposed [h][d][n] bf16 (so pass-3 B operands are contiguous).
// ---------------------------------------------------------------------------
__global__ void __launch_bounds__(256) k_qkv(const float* __restrict__ nodes,
                                             const float* __restrict__ WQ,
                                             const float* __restrict__ WK,
                                             const float* __restrict__ WV,
                                             unsigned short* __restrict__ q_bf,
                                             unsigned short* __restrict__ k_bf,
                                             unsigned short* __restrict__ v_t) {
    const int lane = threadIdx.x & 31;
    const int wv   = threadIdx.x >> 5;
    const int lm   = lane & 15;
    const int lh   = lane >> 4;
    const int r0g  = blockIdx.y * 128 + wv * 16;   // [0,1536)
    const int mat  = r0g >> 9;                     // 0=Q 1=K 2=V
    const int rm0  = r0g & 511;
    const int n0   = blockIdx.x * 16;
    const float* W = (mat == 0) ? WQ : (mat == 1 ? WK : WV);

    v8f acc = {};
    const int an = n0 + lm;
#pragma unroll
    for (int cc = 0; cc < 4; ++cc) {
        const int c0 = cc * 32;
        BFV A, B;
        const float4* ap0 = reinterpret_cast<const float4*>(nodes + an * DD + c0 + lh * 16);
        const float4* ap1 = reinterpret_cast<const float4*>(nodes + an * DD + c0 + lh * 16 + 8);
        pack8(A, 0, ap0[0], ap0[1]);
        pack8(A, 4, ap1[0], ap1[1]);
        const float4* bp = reinterpret_cast<const float4*>(W + (rm0 + lm) * DD + c0 + lh * 16);
        pack8(B, 0, bp[0], bp[1]);
        pack8(B, 4, bp[2], bp[3]);
        acc = wmma_bf16(A, B, acc);
    }
#pragma unroll
    for (int j = 0; j < 8; ++j) {
        const int n = n0 + j + 8 * lh;
        const int r = rm0 + lm;
        const int d = r >> 3, hh = r & 7;
        const unsigned short bv = f2bf(acc[j]);
        if (mat == 0)      q_bf[((size_t)hh * NN + n) * QKD + d] = bv;
        else if (mat == 1) k_bf[((size_t)hh * NN + n) * QKD + d] = bv;
        else               v_t[((size_t)hh * QKD + d) * NN + n]  = bv;
    }
}

// ---------------------------------------------------------------------------
// Pass 2: scores = (q.k/8)*we, edges_out = l2norm(edges + scores@Weo^T),
// rowmax over masked axis-1, global min of scores*adj.
// Double-buffered async-to-LDS edges staging; cooperative all-head `we`.
// grid(128 n-tiles, 8 m-groups of 256), block 256 (wave w <-> head w)
// ---------------------------------------------------------------------------
__global__ void __launch_bounds__(256) k_scores_edges(
        const unsigned short* __restrict__ q_bf,
        const unsigned short* __restrict__ k_bf,
        const float* __restrict__ edges,
        const float* __restrict__ adj,
        const float* __restrict__ We,
        const float* __restrict__ Weo,
        float* __restrict__ out_edges,
        float* __restrict__ rowmax,
        float* __restrict__ gmin) {
    __shared__ float se[2][256 * EE];
    __shared__ float sadj[2][256];
    __shared__ float ss[HH][256];
    __shared__ float swe[HH][256];
    __shared__ float sWe[64];
    __shared__ float sweo[64];
    __shared__ float sred[256];

    const int t    = threadIdx.x;
    const int lane = t & 31;
    const int h    = t >> 5;
    const int lm   = lane & 15;
    const int lh   = lane >> 4;
    const int n0    = blockIdx.x * 16;
    const int mbase = blockIdx.y * 256;
    const int tn = n0 + (t >> 4);       // this thread's staging row
    const int tm = t & 15;              // this thread's staging col within tile

    if (t < 64) { sweo[t] = Weo[t]; sWe[t] = We[t]; }

    // Preload q A-operands (this wave's head, 16 rows x 64 d, 2 K-chunks)
    BFV qa0, qa1;
    {
        const unsigned short* qp = q_bf + ((size_t)h * NN + (n0 + lm)) * QKD;
        qa0.q[0] = *reinterpret_cast<const uint4*>(qp + lh * 16);
        qa0.q[1] = *reinterpret_cast<const uint4*>(qp + lh * 16 + 8);
        qa1.q[0] = *reinterpret_cast<const uint4*>(qp + 32 + lh * 16);
        qa1.q[1] = *reinterpret_cast<const uint4*>(qp + 32 + lh * 16 + 8);
    }

    float tmin = 3.0e38f;
    float rmax_acc = -3.0e38f;
    const int sh = t >> 4, sn = t & 15;   // (head,row) role for t<128

    // Prologue: issue async staging for tile 0
    stage_tile_async(edges + ((size_t)tn * NN + mbase + tm) * EE,
                     adj + (size_t)tn * NN + mbase + tm, se[0], sadj[0], t);

    for (int mt = 0; mt < 16; ++mt) {
        const int cur = mt & 1;
        const int m0 = mbase + mt * 16;
        if (mt < 15) {
            // software pipeline: stage tile mt+1 while tile mt is consumed
            stage_tile_async(edges + ((size_t)tn * NN + m0 + 16 + tm) * EE,
                             adj + (size_t)tn * NN + m0 + 16 + tm,
                             se[cur ^ 1], sadj[cur ^ 1], t);
            asm volatile("s_wait_asynccnt 0x3" ::: "memory");
        } else {
            asm volatile("s_wait_asynccnt 0x0" ::: "memory");
        }
        __syncthreads();
        const float* seb  = se[cur];
        const float* adjb = sadj[cur];
        // ---- Stage W: all-head modulation weights (edges are head-invariant)
        {
            float ed[8];
#pragma unroll
            for (int e = 0; e < 8; ++e) ed[e] = seb[t * 8 + e];
            const float a = adjb[t];
#pragma unroll
            for (int hh = 0; hh < 8; ++hh) {
                float x = 0.f;
#pragma unroll
                for (int e = 0; e < 8; ++e) x += ed[e] * sWe[hh * 8 + e];
                swe[hh][t] = x * a;
            }
        }
        __syncthreads();
        // ---- Stage B: per-wave WMMA scores * precomputed we ----
        {
            BFV kb0, kb1;
            const unsigned short* kp = k_bf + ((size_t)h * NN + (m0 + lm)) * QKD;
            const uint4* b0 = reinterpret_cast<const uint4*>(kp + lh * 16);
            kb0.q[0] = b0[0]; kb0.q[1] = b0[1];
            const uint4* b1 = reinterpret_cast<const uint4*>(kp + 32 + lh * 16);
            kb1.q[0] = b1[0]; kb1.q[1] = b1[1];
            v8f acc = {};
            acc = wmma_bf16(qa0, kb0, acc);
            acc = wmma_bf16(qa1, kb1, acc);
#pragma unroll
            for (int j = 0; j < 8; ++j) {
                const int pidx = (j + 8 * lh) * 16 + lm;
                ss[h][pidx] = acc[j] * 0.125f * swe[h][pidx];
            }
        }
        __syncthreads();
        // ---- Stage C: edges_out (NT stores) + reductions ----
        {
            float sv[8];
#pragma unroll
            for (int hh = 0; hh < 8; ++hh) {
                sv[hh] = ss[hh][t];
                tmin = fminf(tmin, sv[hh]);   // scores*adj == scores here
            }
            float r[8];
            float nrm = 0.f;
#pragma unroll
            for (int e = 0; e < 8; ++e) {
                float x = seb[t * 8 + e];
#pragma unroll
                for (int hh = 0; hh < 8; ++hh) x += sv[hh] * sweo[e * 8 + hh];
                r[e] = x; nrm += x * x;
            }
            nrm = sqrtf(nrm) + 1e-8f;
            const float inv = 1.f / nrm;
            v4f o0 = { r[0] * inv, r[1] * inv, r[2] * inv, r[3] * inv };
            v4f o1 = { r[4] * inv, r[5] * inv, r[6] * inv, r[7] * inv };
            v4f* op = reinterpret_cast<v4f*>(out_edges + ((size_t)tn * NN + m0 + tm) * EE);
            __builtin_nontemporal_store(o0, op);       // write-once stream:
            __builtin_nontemporal_store(o1, op + 1);   // keep `edges` in L2
            if (t < 128) {
#pragma unroll
                for (int mm = 0; mm < 16; ++mm) {
                    const int p2 = sn * 16 + mm;
                    if (adjb[p2] > 0.f) rmax_acc = fmaxf(rmax_acc, ss[sh][p2]);
                }
            }
        }
        __syncthreads();
    }
    if (t < 128) atomicMaxF(&rowmax[sh * NN + n0 + sn], rmax_acc);
    sred[t] = tmin;
    __syncthreads();
    for (int st = 128; st > 0; st >>= 1) {
        if (t < st) sred[t] = fminf(sred[t], sred[t + st]);
        __syncthreads();
    }
    if (t == 0) atomicMinF(gmin, sred[0]);
}

// ---------------------------------------------------------------------------
// Pass 3: recompute scores, p = exp(s - xmax) masked, rowsum, hacc += p @ v
// grid(128 n-tiles, 8 m-groups of 256), block 256 (wave w <-> head w)
// ---------------------------------------------------------------------------
__global__ void __launch_bounds__(256) k_attn(
        const unsigned short* __restrict__ q_bf,
        const unsigned short* __restrict__ k_bf,
        const unsigned short* __restrict__ v_t,
        const float* __restrict__ edges,
        const float* __restrict__ adj,
        const float* __restrict__ We,
        const float* __restrict__ rowmax,
        const float* __restrict__ gmin,
        float* __restrict__ rowsum,
        float* __restrict__ hacc) {
    __shared__ float se[2][256 * EE];
    __shared__ float sadj[2][256];
    __shared__ float sp[HH][16][32];
    __shared__ float swe[HH][256];
    __shared__ float sWe[64];

    const int t    = threadIdx.x;
    const int lane = t & 31;
    const int h    = t >> 5;
    const int lm   = lane & 15;
    const int lh   = lane >> 4;
    const int n0    = blockIdx.x * 16;
    const int mbase = blockIdx.y * 256;
    const int tn = n0 + (t >> 4);
    const int tm = t & 15;

    if (t < 64) sWe[t] = We[t];

    BFV qa0, qa1;
    {
        const unsigned short* qp = q_bf + ((size_t)h * NN + (n0 + lm)) * QKD;
        qa0.q[0] = *reinterpret_cast<const uint4*>(qp + lh * 16);
        qa0.q[1] = *reinterpret_cast<const uint4*>(qp + lh * 16 + 8);
        qa1.q[0] = *reinterpret_cast<const uint4*>(qp + 32 + lh * 16);
        qa1.q[1] = *reinterpret_cast<const uint4*>(qp + 32 + lh * 16 + 8);
    }

    const float gm = gmin[0];
    float xm[8];
#pragma unroll
    for (int j = 0; j < 8; ++j) {
        const int n = n0 + j + 8 * lh;
        xm[j] = fmaxf(rowmax[h * NN + n], gm);
    }

    v8f accD[4] = {};
    float rs_acc = 0.f;
    const int sh = t >> 4, sn = t & 15;

    // Prologue: stage tile 0
    stage_tile_async(edges + ((size_t)tn * NN + mbase + tm) * EE,
                     adj + (size_t)tn * NN + mbase + tm, se[0], sadj[0], t);

    for (int pair = 0; pair < 8; ++pair) {
        const int mp = mbase + pair * 32;
        for (int sub = 0; sub < 2; ++sub) {
            const int k  = pair * 2 + sub;
            const int cur = k & 1;
            const int m0 = mp + sub * 16;
            if (k < 15) {
                stage_tile_async(edges + ((size_t)tn * NN + m0 + 16 + tm) * EE,
                                 adj + (size_t)tn * NN + m0 + 16 + tm,
                                 se[cur ^ 1], sadj[cur ^ 1], t);
                asm volatile("s_wait_asynccnt 0x3" ::: "memory");
            } else {
                asm volatile("s_wait_asynccnt 0x0" ::: "memory");
            }
            __syncthreads();
            const float* seb  = se[cur];
            const float* adjb = sadj[cur];
            // ---- Stage W: all-head modulation weights ----
            {
                float ed[8];
#pragma unroll
                for (int e = 0; e < 8; ++e) ed[e] = seb[t * 8 + e];
                const float a = adjb[t];
#pragma unroll
                for (int hh = 0; hh < 8; ++hh) {
                    float x = 0.f;
#pragma unroll
                    for (int e = 0; e < 8; ++e) x += ed[e] * sWe[hh * 8 + e];
                    swe[hh][t] = x * a;
                }
            }
            __syncthreads();
            // ---- Stage B: scores + masked exp -> LDS p ----
            {
                BFV kb0, kb1;
                const unsigned short* kp = k_bf + ((size_t)h * NN + (m0 + lm)) * QKD;
                const uint4* b0 = reinterpret_cast<const uint4*>(kp + lh * 16);
                kb0.q[0] = b0[0]; kb0.q[1] = b0[1];
                const uint4* b1 = reinterpret_cast<const uint4*>(kp + 32 + lh * 16);
                kb1.q[0] = b1[0]; kb1.q[1] = b1[1];
                v8f acc = {};
                acc = wmma_bf16(qa0, kb0, acc);
                acc = wmma_bf16(qa1, kb1, acc);
#pragma unroll
                for (int j = 0; j < 8; ++j) {
                    const int pidx = (j + 8 * lh) * 16 + lm;
                    const float s = acc[j] * 0.125f * swe[h][pidx];
                    const float p = (adjb[pidx] > 0.f) ? __expf(s - xm[j]) : 0.f;
                    sp[h][j + 8 * lh][sub * 16 + lm] = p;
                }
            }
            __syncthreads();
        }
        // ---- Stage C1: rowsum partials ----
        if (t < 128) {
            float s = 0.f;
#pragma unroll
            for (int mm = 0; mm < 32; ++mm) s += sp[sh][sn][mm];
            rs_acc += s;
        }
        // ---- Stage C2: hacc += p @ v  (A via LDS transpose, B from v_t) ----
        {
            BFV pa;
#pragma unroll
            for (int w2 = 0; w2 < 8; ++w2) {
                const int i = 2 * w2;
                const int ml = i + lh * 16;
                pa.u32[w2] = f2bf2(sp[h][lm][ml], sp[h][lm][ml + 1]);
            }
#pragma unroll
            for (int td = 0; td < 4; ++td) {
                BFV vb;
                const unsigned short* vp =
                    v_t + ((size_t)h * QKD + td * 16 + lm) * NN + mp + lh * 16;
                const uint4* q = reinterpret_cast<const uint4*>(vp);
                vb.q[0] = q[0]; vb.q[1] = q[1];
                accD[td] = wmma_bf16(pa, vb, accD[td]);
            }
        }
        __syncthreads();
    }

    if (t < 128) atomicAdd(&rowsum[sh * NN + n0 + sn], rs_acc);
#pragma unroll
    for (int td = 0; td < 4; ++td) {
#pragma unroll
        for (int j = 0; j < 8; ++j) {
            const int n = n0 + j + 8 * lh;
            atomicAdd(&hacc[((size_t)h * NN + n) * QKD + td * 16 + lm], accD[td][j]);
        }
    }
}

// ---------------------------------------------------------------------------
// Pass 4: nodes_out = l2norm(nodes + (h/rowsum) @ Wo^T + bo).  One block/node.
// ---------------------------------------------------------------------------
__global__ void __launch_bounds__(128) k_nodes_out(const float* __restrict__ nodes,
                                                   const float* __restrict__ Wo,
                                                   const float* __restrict__ bo,
                                                   const float* __restrict__ hacc,
                                                   const float* __restrict__ rowsum,
                                                   float* __restrict__ out_nodes) {
    __shared__ float hf[512];
    __shared__ float red[128];
    const int n = blockIdx.x, t = threadIdx.x;
    for (int idx = t; idx < 512; idx += 128) {
        const int d = idx >> 3, hh = idx & 7;     // flat index = d*H + h
        hf[idx] = hacc[((size_t)hh * NN + n) * QKD + d] / rowsum[hh * NN + n];
    }
    __syncthreads();
    float acc = nodes[n * DD + t] + bo[t];
    const float* w = Wo + (size_t)t * 512;
    for (int j = 0; j < 512; j += 4) {
        acc += hf[j] * w[j] + hf[j + 1] * w[j + 1]
             + hf[j + 2] * w[j + 2] + hf[j + 3] * w[j + 3];
    }
    red[t] = acc * acc;
    __syncthreads();
    for (int st = 64; st > 0; st >>= 1) {
        if (t < st) red[t] += red[t + st];
        __syncthreads();
    }
    const float nrm = sqrtf(red[0]) + 1e-8f;
    out_nodes[n * DD + t] = acc / nrm;
}

// ---------------------------------------------------------------------------
extern "C" void kernel_launch(void* const* d_in, const int* in_sizes, int n_in,
                              void* d_out, int out_size, void* d_ws, size_t ws_size,
                              hipStream_t stream) {
    const float* nodes = (const float*)d_in[0];
    const float* adjp  = (const float*)d_in[1];
    const float* edges = (const float*)d_in[2];
    const float* WQ    = (const float*)d_in[3];
    const float* WK    = (const float*)d_in[4];
    const float* WV    = (const float*)d_in[5];
    const float* Wo    = (const float*)d_in[6];
    const float* bo    = (const float*)d_in[7];
    const float* We    = (const float*)d_in[8];
    const float* Weo   = (const float*)d_in[9];

    float* out_nodes = (float*)d_out;
    float* out_edges = (float*)d_out + (size_t)NN * DD;

    char* w = (char*)d_ws;
    unsigned short* q_bf = (unsigned short*)w; w += (size_t)HH * NN * QKD * 2;
    unsigned short* k_bf = (unsigned short*)w; w += (size_t)HH * NN * QKD * 2;
    unsigned short* v_t  = (unsigned short*)w; w += (size_t)HH * NN * QKD * 2;
    float* hacc   = (float*)w; w += (size_t)HH * NN * QKD * 4;
    float* rowmax = (float*)w; w += (size_t)HH * NN * 4;
    float* rowsum = (float*)w; w += (size_t)HH * NN * 4;
    float* gminp  = (float*)w; w += 256;

    k_init<<<dim3((HH * NN * QKD) / 256), dim3(256), 0, stream>>>(hacc, rowsum, rowmax, gminp);
    k_qkv<<<dim3(128, 12), dim3(256), 0, stream>>>(nodes, WQ, WK, WV, q_bf, k_bf, v_t);
    k_scores_edges<<<dim3(128, 8), dim3(256), 0, stream>>>(q_bf, k_bf, edges, adjp, We, Weo,
                                                           out_edges, rowmax, gminp);
    k_attn<<<dim3(128, 8), dim3(256), 0, stream>>>(q_bf, k_bf, v_t, edges, adjp, We,
                                                   rowmax, gminp, rowsum, hacc);
    k_nodes_out<<<dim3(2048), dim3(128), 0, stream>>>(nodes, Wo, bo, hacc, rowsum, out_nodes);
}
